// EmbeddingDAE_30477087932478
// MI455X (gfx1250) — matrix operand order repaired and checked by reading, not compiled, not run on hardware
//
#include <hip/hip_runtime.h>
#include <hip/hip_bf16.h>

typedef __attribute__((ext_vector_type(16))) _Float16 v16h;
typedef __attribute__((ext_vector_type(8)))  _Float16 v8h;
typedef __attribute__((ext_vector_type(8)))  float    v8f;

#define B_   4096
#define L_   200
#define C_   500
#define D_   128
#define HPIT 136   // activation-tile row pitch in halfs (272 B = 17*16 -> rows stay 16B aligned)

static constexpr float LN_EPS = 1e-5f;

// ---------------------------------------------------------------------------
// Kernel 1: ragged masked mean-pool.  One wave per batch row; lane holds 4 dims.
// ---------------------------------------------------------------------------
__global__ void __launch_bounds__(256)
pool_kernel(const float* __restrict__ emb,
            const int*   __restrict__ remain_ids,
            const int*   __restrict__ remain_len,
            float*       __restrict__ p0)
{
    const int wave = threadIdx.x >> 5;
    const int lane = threadIdx.x & 31;
    const int b    = blockIdx.x * 8 + wave;
    if (b >= B_) return;

    const int len = remain_len[b];
    const int* ids = remain_ids + (size_t)b * L_;
    float4 acc = make_float4(0.f, 0.f, 0.f, 0.f);
    for (int l = 0; l < len; ++l) {
        const int id = ids[l];
        const float4 v = ((const float4*)(emb + (size_t)id * D_))[lane];
        acc.x += v.x; acc.y += v.y; acc.z += v.z; acc.w += v.w;
    }
    const float inv = 1.0f / (float)(len > 1 ? len : 1);
    acc.x *= inv; acc.y *= inv; acc.z *= inv; acc.w *= inv;
    ((float4*)(p0 + (size_t)b * D_))[lane] = acc;
}

// ---------------------------------------------------------------------------
// WMMA fragment helpers (ISA 7.12.2 layouts, wave32)
// ---------------------------------------------------------------------------
__device__ __forceinline__ v8f wmma_f16(v16h a, v16h b, v8f c) {
    // (neg_a, A, neg_b, B, c_mod, C, reuse_a, reuse_b)
    return __builtin_amdgcn_wmma_f32_16x16x32_f16(false, a, false, b,
                                                  (short)0, c, false, false);
}

// A (16x32 f16): lane's 16 elements are two contiguous 8-half groups in a
// row-major tile: [kc*32+kh .. +7] and [kc*32+16+kh .. +7]  (kh = (lane>=16)*8)
__device__ __forceinline__ v16h load_afrag(const _Float16* hbuf, int lane, int kc) {
    const int row = lane & 15;
    const int kh  = (lane >> 4) << 3;            // 0 or 8
    const _Float16* hr = hbuf + row * HPIT + kc * 32 + kh;
    const v8h g0 = *(const v8h*)(hr);            // 16B aligned -> ds_load_b128
    const v8h g1 = *(const v8h*)(hr + 16);       // 16B aligned -> ds_load_b128
    return __builtin_shufflevector(g0, g1, 0, 1, 2, 3, 4, 5, 6, 7,
                                           8, 9, 10, 11, 12, 13, 14, 15);
}

// B (32x16 f16): weights pre-swizzled at staging time so each (kc, t, lane)
// fragment is one contiguous 32B run -> a single v16h load (2x ds_load_b128).
__device__ __forceinline__ v16h load_bfrag(const _Float16* wswz, int lane, int kc, int t) {
    return *(const v16h*)(wswz + ((((kc << 3) + t) << 5) + lane) * 16);
}

// stage W (f32 row-major [k][n]) -> LDS in B-fragment order
__device__ __forceinline__ void stage_w(const float* __restrict__ W, _Float16* wswz) {
    for (int i = threadIdx.x; i < D_ * D_; i += 256) {
        const int k = i >> 7, n = i & 127;
        const int kc = k >> 5, kin = k & 31;
        const int khalf = (kin >> 3) & 1, e = (kin & 7) | ((kin >> 4) << 3);   // lane half & element index
        const int t = n >> 4, col = n & 15;
        const int lane = (khalf << 4) | col;
        wswz[((((kc << 3) + t) << 5) + lane) * 16 + e] = (_Float16)W[i];
    }
}

// ---------------------------------------------------------------------------
// Kernel 2: MLP (2x GEMM+bias+ReLU) + LayerNorm + empty-playlist mask.
// 256 threads = 8 waves; each wave owns a 16-row tile of the batch.
// ---------------------------------------------------------------------------
__global__ void __launch_bounds__(256)
mlp_ln_kernel(const float* __restrict__ p0,
              const float* __restrict__ W1, const float* __restrict__ b1,
              const float* __restrict__ W2, const float* __restrict__ b2,
              const float* __restrict__ gamma, const float* __restrict__ beta,
              const int*   __restrict__ remain_len,
              float*       __restrict__ pfin)
{
    __shared__ __align__(16) _Float16 Wswz[D_ * D_];       // 32768 B, W1 then W2
    __shared__ __align__(16) _Float16 Hlds[8 * 16 * HPIT]; // 34816 B, per-wave tile

    const int wave = threadIdx.x >> 5;
    const int lane = threadIdx.x & 31;
    const int b0   = (blockIdx.x * 8 + wave) * 16;
    _Float16* hbuf = Hlds + wave * 16 * HPIT;

    const int half = lane >> 4;
    const int col  = lane & 15;

    // ---- stage W1 (swizzled) and pooled rows into LDS ----
    stage_w(W1, Wswz);
    for (int i = lane; i < 16 * D_; i += 32) {
        const int r = i >> 7, n = i & 127;
        hbuf[r * HPIT + n] = (_Float16)p0[(size_t)(b0 + r) * D_ + n];
    }
    __syncthreads();

    // ---- layer 1: h1 = relu(p @ W1 + b1) ----
    {
        v16h a0 = load_afrag(hbuf, lane, 0);
        v16h a1 = load_afrag(hbuf, lane, 1);
        v16h a2 = load_afrag(hbuf, lane, 2);
        v16h a3 = load_afrag(hbuf, lane, 3);
        v8f acc[8];
#pragma unroll
        for (int t = 0; t < 8; ++t) {
            const float bias = b1[t * 16 + col];
            v8f c;
#pragma unroll
            for (int r = 0; r < 8; ++r) c[r] = bias;
            c = wmma_f16(a0, load_bfrag(Wswz, lane, 0, t), c);
            c = wmma_f16(a1, load_bfrag(Wswz, lane, 1, t), c);
            c = wmma_f16(a2, load_bfrag(Wswz, lane, 2, t), c);
            c = wmma_f16(a3, load_bfrag(Wswz, lane, 3, t), c);
#pragma unroll
            for (int r = 0; r < 8; ++r) c[r] = fmaxf(c[r], 0.f);
            acc[t] = c;
        }
        __syncthreads();   // all waves done reading W1 before the rewrite below
        // write h1 tile back to LDS in row-major (C layout -> memory)
#pragma unroll
        for (int t = 0; t < 8; ++t)
#pragma unroll
            for (int r = 0; r < 8; ++r)
                hbuf[(r + half * 8) * HPIT + t * 16 + col] = (_Float16)acc[t][r];
    }
    __syncthreads();

    // ---- stage W2 (swizzled) ----
    stage_w(W2, Wswz);
    __syncthreads();

    // ---- layer 2 + LayerNorm + mask ----
    {
        v16h a0 = load_afrag(hbuf, lane, 0);
        v16h a1 = load_afrag(hbuf, lane, 1);
        v16h a2 = load_afrag(hbuf, lane, 2);
        v16h a3 = load_afrag(hbuf, lane, 3);
        v8f acc[8];
#pragma unroll
        for (int t = 0; t < 8; ++t) {
            const float bias = b2[t * 16 + col];
            v8f c;
#pragma unroll
            for (int r = 0; r < 8; ++r) c[r] = bias;
            c = wmma_f16(a0, load_bfrag(Wswz, lane, 0, t), c);
            c = wmma_f16(a1, load_bfrag(Wswz, lane, 1, t), c);
            c = wmma_f16(a2, load_bfrag(Wswz, lane, 2, t), c);
            c = wmma_f16(a3, load_bfrag(Wswz, lane, 3, t), c);
#pragma unroll
            for (int r = 0; r < 8; ++r) c[r] = fmaxf(c[r], 0.f);
            acc[t] = c;
        }

        // row-wise sums for LayerNorm: reduce over 8 n-tiles then 16 lanes/half
        float sr[8], qr[8];
#pragma unroll
        for (int r = 0; r < 8; ++r) {
            float s = 0.f, q = 0.f;
#pragma unroll
            for (int t = 0; t < 8; ++t) { const float x = acc[t][r]; s += x; q += x * x; }
            sr[r] = s; qr[r] = q;
        }
#pragma unroll
        for (int m = 1; m <= 8; m <<= 1) {
#pragma unroll
            for (int r = 0; r < 8; ++r) {
                sr[r] += __shfl_xor(sr[r], m, 32);
                qr[r] += __shfl_xor(qr[r], m, 32);
            }
        }

        float gcol[8], bcol[8];
#pragma unroll
        for (int t = 0; t < 8; ++t) { gcol[t] = gamma[t * 16 + col]; bcol[t] = beta[t * 16 + col]; }

#pragma unroll
        for (int r = 0; r < 8; ++r) {
            const float mu  = sr[r] * (1.f / (float)D_);
            const float var = qr[r] * (1.f / (float)D_) - mu * mu;
            const float rs  = rsqrtf(var + LN_EPS);
            const int brow  = b0 + r + half * 8;
            const float zf  = (remain_len[brow] > 0) ? 1.f : 0.f;
            float* outrow = pfin + (size_t)brow * D_;
#pragma unroll
            for (int t = 0; t < 8; ++t) {
                const float x = (acc[t][r] - mu) * rs * gcol[t] + bcol[t];
                outrow[t * 16 + col] = x * zf;
            }
        }
    }
}

// ---------------------------------------------------------------------------
// Kernel 3: candidate scoring logits[b,c] = emb[cand_ids[b,c]] . p[b]
// One block per batch row; one wave per candidate; float4 loads + shfl reduce.
// ---------------------------------------------------------------------------
__global__ void __launch_bounds__(256)
score_kernel(const float* __restrict__ emb,
             const float* __restrict__ pfin,
             const int*   __restrict__ cand_ids,
             float*       __restrict__ out)
{
    __shared__ float pl[D_];
    const int b = blockIdx.x;
    if (threadIdx.x < D_) pl[threadIdx.x] = pfin[(size_t)b * D_ + threadIdx.x];
    __syncthreads();

    const int wave = threadIdx.x >> 5;
    const int lane = threadIdx.x & 31;
    const float4 pv = ((const float4*)pl)[lane];
    const int* cids = cand_ids + (size_t)b * C_;

    for (int c = wave; c < C_; c += 8) {
        const int id = cids[c];
        if (c + 8 < C_) {
            const int nid = cids[c + 8];
            __builtin_prefetch(emb + (size_t)nid * D_ + lane * 4, 0, 0);
        }
        const float4 ev = ((const float4*)(emb + (size_t)id * D_))[lane];
        float d = ev.x * pv.x + ev.y * pv.y + ev.z * pv.z + ev.w * pv.w;
#pragma unroll
        for (int m = 16; m >= 1; m >>= 1) d += __shfl_xor(d, m, 32);
        if (lane == 0) out[(size_t)b * C_ + c] = d;
    }
}

// ---------------------------------------------------------------------------
extern "C" void kernel_launch(void* const* d_in, const int* in_sizes, int n_in,
                              void* d_out, int out_size, void* d_ws, size_t ws_size,
                              hipStream_t stream)
{
    const float* emb        = (const float*)d_in[0];
    const float* W1         = (const float*)d_in[1];
    const float* b1         = (const float*)d_in[2];
    const float* W2         = (const float*)d_in[3];
    const float* b2         = (const float*)d_in[4];
    const float* gamma      = (const float*)d_in[5];
    const float* beta       = (const float*)d_in[6];
    const int*   remain_ids = (const int*)d_in[7];
    const int*   remain_len = (const int*)d_in[8];
    const int*   cand_ids   = (const int*)d_in[9];
    float*       out        = (float*)d_out;

    float* p0   = (float*)d_ws;                 // [B, D] pooled
    float* pfin = p0 + (size_t)B_ * D_;         // [B, D] encoded (4 MB total scratch)

    pool_kernel  <<<B_ / 8,   256, 0, stream>>>(emb, remain_ids, remain_len, p0);
    mlp_ln_kernel<<<B_ / 128, 256, 0, stream>>>(p0, W1, b1, W2, b2, gamma, beta,
                                                remain_len, pfin);
    score_kernel <<<B_,       256, 0, stream>>>(emb, pfin, cand_ids, out);
}
